// DependencyGNN_7421703487729
// MI455X (gfx1250) — matrix operand order repaired and checked by reading, not compiled, not run on hardware
//
#include <hip/hip_runtime.h>

namespace {

constexpr int B_ = 8, L_ = 256, H_ = 768, NH_ = 12, HD_ = 64, IT_ = 3, NLAB_ = 50, C_ = 768;
constexpr int LP1_ = L_ + 1;                 // 257 (relay + nodes)
constexpr float SCALE_ = 0.125f;             // 1/sqrt(64)
constexpr float EPS_ = 1e-5f;
constexpr float SLOPE_ = 0.01f;              // leaky_relu default

typedef __attribute__((ext_vector_type(16))) __bf16 bf16x16;
typedef __attribute__((ext_vector_type(8)))  float  f32x8;

// ---------------------------------------------------------------------------
// Block reductions (blockDim.x == 256)
// ---------------------------------------------------------------------------
__device__ __forceinline__ float block_sum(float v, float* red) {
  int t = threadIdx.x;
  red[t] = v; __syncthreads();
  for (int s = 128; s > 0; s >>= 1) { if (t < s) red[t] += red[t + s]; __syncthreads(); }
  float r = red[0]; __syncthreads();
  return r;
}
__device__ __forceinline__ float block_max(float v, float* red) {
  int t = threadIdx.x;
  red[t] = v; __syncthreads();
  for (int s = 128; s > 0; s >>= 1) { if (t < s) red[t] = fmaxf(red[t], red[t + s]); __syncthreads(); }
  float r = red[0]; __syncthreads();
  return r;
}

// ---------------------------------------------------------------------------
// CDNA5 async global->LDS copy (ASYNCcnt-tracked), inline asm
// ---------------------------------------------------------------------------
__device__ __forceinline__ unsigned lds_off32(const void* p) {
  // flat LDS aperture: addr[31:0] is the wave-relative LDS byte offset
  return (unsigned)(uintptr_t)p;
}
__device__ __forceinline__ void async_b128(const __bf16* gaddr, unsigned ldsaddr) {
  asm volatile("global_load_async_to_lds_b128 %0, %1, off" :: "v"(ldsaddr), "v"(gaddr) : "memory");
}
__device__ __forceinline__ void wait_async_le3() { asm volatile("s_wait_asynccnt 0x3" ::: "memory"); }
__device__ __forceinline__ void wait_async_le2() { asm volatile("s_wait_asynccnt 0x2" ::: "memory"); }
__device__ __forceinline__ void wait_async_le0() { asm volatile("s_wait_asynccnt 0x0" ::: "memory"); }

// ---------------------------------------------------------------------------
// WMMA bf16 fragment loaders from LDS tiles (contiguous k = 16*half + s per lane, wave32)
// ---------------------------------------------------------------------------
// A tile [64][32] row-major; fragment rows mi*16..mi*16+15.
__device__ __forceinline__ bf16x16 frag_a_lds(const __bf16* At, int mi) {
  int lane = threadIdx.x & 31;
  int half = lane >> 4;
  const __bf16* p = At + (mi * 16 + (lane & 15)) * 32;
  bf16x16 a;
#pragma unroll
  for (int v = 0; v < 8; ++v) {
    int kb = 2 * v + half * 16;
    a[2 * v]     = p[kb];
    a[2 * v + 1] = p[kb + 1];
  }
  return a;
}
// W tile [BN][32] (n-major, A@W^T): 16 contiguous K per lane.
__device__ __forceinline__ bf16x16 frag_bT_lds(const __bf16* Bt, int n0) {
  int lane = threadIdx.x & 31;
  const __bf16* p = Bt + (n0 + (lane & 15)) * 32 + ((lane >> 4) << 4);
  bf16x16 b;
#pragma unroll
  for (int v = 0; v < 16; ++v) b[v] = p[v];
  return b;
}
// B tile [32][64] (k-major): strided per-lane loads.
__device__ __forceinline__ bf16x16 frag_b_lds(const __bf16* Bt, int n0) {
  int lane = threadIdx.x & 31;
  int col = n0 + (lane & 15);
  int kb = (lane >> 4) << 4;
  bf16x16 b;
#pragma unroll
  for (int v = 0; v < 8; ++v) {
    b[2 * v]     = Bt[(kb + 2 * v) * 64 + col];
    b[2 * v + 1] = Bt[(kb + 2 * v + 1) * 64 + col];
  }
  return b;
}

// ---------------------------------------------------------------------------
// Double-buffered async-LDS GEMM mainloop. Block = 256 thr (8 waves, 4x2).
// Block tile 64(M) x BN(N), K step 32. BT: B operand is W[N,K] (A @ W^T,
// BN=64 or 128); !BT: B[K,N], BN=64. A rows clamped (masked at store).
// LDS layout (bf16 elems): A dbl @ [0,4096), B dbl @ [4096, 4096+2*BS).
// ---------------------------------------------------------------------------
template <bool BT, int BN>
__device__ __forceinline__ void gemm_mainloop(const __bf16* __restrict__ Ab, int lda, int mclamp,
                                              const __bf16* __restrict__ Bb, int ldb, int K,
                                              int m0blk, int n0blk, __bf16* lds,
                                              f32x8 (&acc)[BN / 32]) {
  constexpr int BS = BT ? (BN * 32) : 2048;       // B tile elems per buffer
  constexpr int NB = BN / 32;                     // 16-wide frags per wave
  const int tid = threadIdx.x;
  const int wave = tid >> 5;
  const int mi = wave & 3, ni = wave >> 2;

  // cooperative staging coords: one b128 (8 bf16) per thread per issue
  const int arow = tid >> 2, acol = (tid & 3) * 8;          // A tile 64x32
  int arg = m0blk + arow; if (arg > mclamp) arg = mclamp;   // clamp, no exec diamonds
  const int brow = BT ? (tid >> 2) : (tid >> 3);            // W rows / B k-rows
  const int bcol = BT ? ((tid & 3) * 8) : ((tid & 7) * 8);

  const __bf16* aG  = Ab + (long)arg * lda + acol;
  const __bf16* bG0 = BT ? (Bb + (long)(n0blk + brow) * ldb + bcol)
                         : (Bb + (long)brow * ldb + n0blk + bcol);
  const __bf16* bG1 = (BT && BN == 128) ? (Bb + (long)(n0blk + brow + 64) * ldb + bcol) : bG0;

  const unsigned base  = lds_off32(lds);
  const unsigned aLoff = (unsigned)(arow * 32 + acol) * 2u;
  const unsigned bLoff = (unsigned)(BT ? (brow * 32 + bcol) : (brow * 64 + bcol)) * 2u;

  auto issue = [&](int k0, int bsel) {
    async_b128(aG + k0, base + (unsigned)bsel * 4096u + aLoff);
    unsigned bbase = base + 8192u + (unsigned)bsel * (unsigned)(BS * 2);
    if (BT) {
      async_b128(bG0 + k0, bbase + bLoff);
      if (BN == 128) async_b128(bG1 + k0, bbase + (unsigned)(64 * 32 * 2) + bLoff);
    } else {
      async_b128(bG0 + (long)k0 * ldb, bbase + bLoff);
    }
  };
  auto compute = [&](int bsel) {
    const __bf16* At = lds + bsel * 2048;
    const __bf16* Bt = lds + 4096 + bsel * BS;
    bf16x16 a = frag_a_lds(At, mi);
#pragma unroll
    for (int t = 0; t < NB; ++t) {
      bf16x16 bfr;
      if (BT) bfr = frag_bT_lds(Bt, ni * (BN / 2) + 16 * t);
      else    bfr = frag_b_lds (Bt, ni * 32 + 16 * t);
      acc[t] = __builtin_amdgcn_wmma_f32_16x16x32_bf16(false, a, false, bfr, (short)0,
                                                       acc[t], false, false);
    }
  };

  issue(0, 0);
  int buf = 0;
  for (int k0 = 0; k0 < K - 32; k0 += 32) {     // branch-free steady state
    issue(k0 + 32, buf ^ 1);
    if constexpr (BT && BN == 128) wait_async_le3(); else wait_async_le2();
    __syncthreads();
    compute(buf);
    __syncthreads();
    buf ^= 1;
  }
  wait_async_le0();                             // peeled tail
  __syncthreads();
  compute(buf);
}

// Generic batched GEMM: D[b] = A[b] @ W^T + bias ; f32 and/or bf16 outputs.
// Block tile 64x128, grid = (N/128, ceil(M/64), batch).
__global__ __launch_bounds__(256)
void gemm_awt_kernel(const __bf16* __restrict__ A, long sA,
                     const __bf16* __restrict__ W,
                     const float*  __restrict__ bias,
                     float*        __restrict__ Df,  long sDf,
                     __bf16*       __restrict__ Dbf, long sDbf,
                     int M, int lda, int K, int ldd) {
  __shared__ __bf16 lds[4096 + 8192];           // 24 KB
  int bz = blockIdx.z;
  int wave = threadIdx.x >> 5, lane = threadIdx.x & 31;
  int mi = wave & 3, ni = wave >> 2;
  int m0blk = blockIdx.y * 64, n0blk = blockIdx.x * 128;
  f32x8 acc[4] = {};
  gemm_mainloop<true, 128>(A + (long)bz * sA, lda, M - 1, W, K, K, m0blk, n0blk, lds, acc);
  int half = lane >> 4, cl = lane & 15;
  int m0 = m0blk + mi * 16, n0 = n0blk + ni * 64;
#pragma unroll
  for (int t = 0; t < 4; ++t) {
    int col = n0 + 16 * t + cl;
    float bv = bias ? bias[col] : 0.f;
#pragma unroll
    for (int r = 0; r < 8; ++r) {
      int row = m0 + r + 8 * half;
      if (row >= M) continue;
      float v = acc[t][r] + bv;
      long bse = (long)row * ldd + col;
      if (Df)  Df [(long)bz * sDf  + bse] = v;
      if (Dbf) Dbf[(long)bz * sDbf + bse] = (__bf16)v;
    }
  }
}

// att = alpha @ q_head per (b, head); epilogue writes the torch-view scramble:
// att[b,n,li,d] -> att2t[b, (li%4)*64 + d, n*64 + li/4]  (bf16)
// M=256, N=64, K=256. grid = (1, 4, B*NH)
__global__ __launch_bounds__(256)
void gemm_att_kernel(const __bf16* __restrict__ alpha,
                     const __bf16* __restrict__ qbf,
                     __bf16*       __restrict__ att2t) {
  __shared__ __bf16 lds[4096 + 4096];           // 16 KB
  int bn = blockIdx.z, b = bn / NH_, n = bn % NH_;
  const __bf16* A  = alpha + (long)bn * L_ * L_;
  const __bf16* Bm = qbf + (long)b * L_ * C_ + n * HD_;
  int wave = threadIdx.x >> 5, lane = threadIdx.x & 31;
  int mi = wave & 3, ni = wave >> 2;
  int m0blk = blockIdx.y * 64;
  f32x8 acc[2] = {};
  gemm_mainloop<false, 64>(A, L_, L_ - 1, Bm, C_, L_, m0blk, 0, lds, acc);
  int half = lane >> 4, cl = lane & 15;
  int m0 = m0blk + mi * 16, n0 = ni * 32;
  __bf16* out = att2t + (long)b * L_ * C_;
#pragma unroll
  for (int r = 0; r < 8; ++r) {
    int li = m0 + r + 8 * half;
    int d0 = n0 + cl, d1 = n0 + 16 + cl;
    int cidx = n * 64 + (li >> 2);
    int llb  = (li & 3) * 64;
    out[(long)(llb + d0) * C_ + cidx] = (__bf16)acc[0][r];
    out[(long)(llb + d1) * C_ + cidx] = (__bf16)acc[1][r];
  }
}

// ---------------------------------------------------------------------------
// Small scalar kernels
// ---------------------------------------------------------------------------
__global__ void copy_f32_kernel(const float* in, float* out, long n) {
  long i = (long)blockIdx.x * blockDim.x + threadIdx.x;
  if (i < n) out[i] = in[i];
}
__global__ void conv_bf16_kernel(const float* in, __bf16* out, long n) {
  long i = (long)blockIdx.x * blockDim.x + threadIdx.x;
  if (i < n) out[i] = (__bf16)in[i];
}
__global__ void relay_init_kernel(const float* data, float* relay) {
  int i = blockIdx.x * blockDim.x + threadIdx.x;       // B*H threads
  if (i >= B_ * H_) return;
  int b = i / H_, h = i % H_;
  float s = 0.f;
  for (int l = 0; l < L_; ++l) s += data[((long)b * L_ + l) * H_ + h];
  relay[i] = s * (1.f / L_);
}

// Fold edge-label machinery into T[label][head] (one block, 256 threads)
__global__ void table_kernel(const float* __restrict__ we, const float* __restrict__ bwe,
                             const float* __restrict__ emb, const float* __restrict__ wedge,
                             float* __restrict__ T) {
  __shared__ float Ms[NH_ * HD_];
  __shared__ float bMs[NH_];
  const float* w3 = wedge + 2 * HD_;
  int t = threadIdx.x;
  for (int idx = t; idx < NH_ * HD_; idx += 256) {
    int n = idx / HD_, e = idx % HD_;
    float s = 0.f;
    for (int d = 0; d < HD_; ++d) s += we[(long)(n * HD_ + d) * HD_ + e] * w3[d];
    Ms[idx] = s;
  }
  if (t < NH_) {
    float s = 0.f;
    for (int d = 0; d < HD_; ++d) s += bwe[t * HD_ + d] * w3[d];
    bMs[t] = s;
  }
  __syncthreads();
  for (int idx = t; idx < NLAB_ * NH_; idx += 256) {
    int lab = idx / NH_, n = idx % NH_;
    float s = bMs[n];
    for (int e = 0; e < HD_; ++e) s += emb[lab * HD_ + e] * Ms[n * HD_ + e];
    T[idx] = s;
  }
}

// LN(x) -> bf16 (one block per row of 768)
__global__ __launch_bounds__(256)
void ln_bf_kernel(const float* __restrict__ x, const float* __restrict__ g,
                  const float* __restrict__ bta, __bf16* __restrict__ out_bf) {
  __shared__ float red[256];
  long row = blockIdx.x;
  const float* xr = x + row * H_;
  int t = threadIdx.x;
  float v[3];
#pragma unroll
  for (int j = 0; j < 3; ++j) v[j] = xr[t + j * 256];
  float mean = block_sum(v[0] + v[1] + v[2], red) * (1.f / H_);
  float d0 = v[0] - mean, d1 = v[1] - mean, d2 = v[2] - mean;
  float var = block_sum(d0 * d0 + d1 * d1 + d2 * d2, red) * (1.f / H_);
  float rstd = rsqrtf(var + EPS_);
#pragma unroll
  for (int j = 0; j < 3; ++j) {
    int h = t + j * 256;
    out_bf[row * H_ + h] = (__bf16)((v[j] - mean) * rstd * g[h] + bta[h]);
  }
}

// nodes = LN(nodes + where(mask,0,leaky(ret))) ; also emit y_bf row (1+l)
__global__ __launch_bounds__(256)
void ln_res_kernel(float* __restrict__ nodes, const float* __restrict__ ret,
                   const int* __restrict__ mask, const float* __restrict__ g,
                   const float* __restrict__ bta, __bf16* __restrict__ y_bf) {
  __shared__ float red[256];
  long row = blockIdx.x;                       // b*L + l
  int b = (int)(row / L_), l = (int)(row % L_);
  bool pad = mask[row] != 0;
  int t = threadIdx.x;
  float v[3];
#pragma unroll
  for (int j = 0; j < 3; ++j) {
    int h = t + j * 256;
    float r = ret[row * H_ + h];
    float lr = (r > 0.f) ? r : SLOPE_ * r;
    v[j] = nodes[row * H_ + h] + (pad ? 0.f : lr);
  }
  float mean = block_sum(v[0] + v[1] + v[2], red) * (1.f / H_);
  float d0 = v[0] - mean, d1 = v[1] - mean, d2 = v[2] - mean;
  float var = block_sum(d0 * d0 + d1 * d1 + d2 * d2, red) * (1.f / H_);
  float rstd = rsqrtf(var + EPS_);
#pragma unroll
  for (int j = 0; j < 3; ++j) {
    int h = t + j * 256;
    float y = (v[j] - mean) * rstd * g[h] + bta[h];
    nodes[row * H_ + h] = y;
    y_bf[((long)b * LP1_ + 1 + l) * C_ + h] = (__bf16)y;
  }
}

__global__ void relay_to_y_kernel(const float* relay, __bf16* y_bf) {
  int i = blockIdx.x * blockDim.x + threadIdx.x;       // B*C
  if (i >= B_ * C_) return;
  int b = i / C_, c = i % C_;
  y_bf[(long)b * LP1_ * C_ + c] = (__bf16)relay[i];
}

// s1/s2: per (b,n,l) 64-dot of q-head with w1/w2
__global__ void s12_kernel(const float* __restrict__ q, const float* __restrict__ wedge,
                           float* __restrict__ s1, float* __restrict__ s2) {
  int i = blockIdx.x * blockDim.x + threadIdx.x;       // B*NH*L
  if (i >= B_ * NH_ * L_) return;
  int b = i / (NH_ * L_), n = (i / L_) % NH_, l = i % L_;
  const float* qv = q + ((long)b * L_ + l) * C_ + n * HD_;
  float a = 0.f, c = 0.f;
  for (int d = 0; d < HD_; ++d) { a += qv[d] * wedge[d]; c += qv[d] * wedge[HD_ + d]; }
  s1[i] = a; s2[i] = c;
}

// One block per (b,i): 12 head softmaxes over j, write alpha bf16
__global__ __launch_bounds__(256)
void edge_softmax_kernel(const float* __restrict__ s1, const float* __restrict__ s2,
                         const float* __restrict__ T, const int* __restrict__ edge_matrix,
                         const int* __restrict__ edge_label, const float* __restrict__ bwedge,
                         float invtemp, __bf16* __restrict__ alpha) {
  __shared__ float red[256];
  int b = blockIdx.x / L_, i = blockIdx.x % L_;
  int j = threadIdx.x;
  long eidx = ((long)b * L_ + i) * L_ + j;
  int  lab  = edge_label[eidx];
  bool edge = edge_matrix[eidx] != 0;
  float bw = *bwedge;
  for (int n = 0; n < NH_; ++n) {
    float sc = edge
             ? (s1[((long)b * NH_ + n) * L_ + i] + s2[((long)b * NH_ + n) * L_ + j]
                + T[lab * NH_ + n] + bw) * SCALE_
             : -1e15f;
    float mx = block_max(sc, red);
    float e  = __expf(sc - mx);
    float sm = block_sum(e, red);
    alpha[(((long)b * NH_ + n) * L_ + i) * L_ + j] = (__bf16)(e / sm * invtemp);
  }
}

// qs = relay @ s_wq^T + s_bq  (tiny: B*C dots of 768)
__global__ void qs_kernel(const float* __restrict__ relay, const float* __restrict__ swq,
                          const float* __restrict__ sbq, float* __restrict__ qs) {
  int i = blockIdx.x * blockDim.x + threadIdx.x;       // B*C
  if (i >= B_ * C_) return;
  int b = i / C_, c = i % C_;
  const float* r = relay + (long)b * H_;
  const float* w = swq + (long)c * H_;
  float s = sbq[c];
  for (int h = 0; h < H_; ++h) s += r[h] * w[h];
  qs[i] = s;
}

// Star attention per (b, head) over 257 slots (v = k)
__global__ __launch_bounds__(256)
void star_kernel(const float* __restrict__ qs, const float* __restrict__ kk,
                 const int* __restrict__ mask, float* __restrict__ sout_pre) {
  __shared__ float pre[LP1_];
  __shared__ float red[256];
  int b = blockIdx.x / NH_, n = blockIdx.x % NH_;
  int t = threadIdx.x;
  const float* qv = qs + (long)b * C_ + n * HD_;
  auto score = [&](int j) -> float {
    const float* kv = kk + ((long)b * LP1_ + j) * C_ + n * HD_;
    float s = 0.f;
    for (int d = 0; d < HD_; ++d) s += qv[d] * kv[d];
    s *= SCALE_;
    if (j >= 1 && mask[b * L_ + (j - 1)] != 0) s = -__builtin_inff();
    return s;
  };
  float p0 = score(t);
  pre[t] = p0;
  float p256 = 0.f;
  if (t == 0) { p256 = score(L_); pre[L_] = p256; }
  __syncthreads();
  float m = (t == 0) ? fmaxf(p0, p256) : p0;
  float mx = block_max(m, red);
  float e = __expf(p0 - mx);
  float e256 = (t == 0) ? __expf(p256 - mx) : 0.f;
  float sm = block_sum(e + e256, red);
  pre[t] = e / sm;
  if (t == 0) pre[L_] = e256 / sm;
  __syncthreads();
  if (t < HD_) {
    float s = 0.f;
    for (int j = 0; j < LP1_; ++j) s += pre[j] * kk[((long)b * LP1_ + j) * C_ + n * HD_ + t];
    sout_pre[(long)b * C_ + n * HD_ + t] = s;
  }
}

// relay = LN(relay + leaky(sout_pre @ s_wo^T + s_bo)), one block per b
__global__ __launch_bounds__(256)
void relay_update_kernel(float* __restrict__ relay, const float* __restrict__ sout_pre,
                         const float* __restrict__ swo, const float* __restrict__ sbo,
                         const float* __restrict__ g, const float* __restrict__ bta) {
  __shared__ float sv[C_];
  __shared__ float red[256];
  int b = blockIdx.x, t = threadIdx.x;
  for (int idx = t; idx < C_; idx += 256) sv[idx] = sout_pre[(long)b * C_ + idx];
  __syncthreads();
  float v[3];
#pragma unroll
  for (int j = 0; j < 3; ++j) {
    int h = t + j * 256;
    float s = sbo[h];
    const float* w = swo + (long)h * C_;
    for (int c = 0; c < C_; ++c) s += w[c] * sv[c];
    float lr = (s > 0.f) ? s : SLOPE_ * s;
    v[j] = relay[(long)b * H_ + h] + lr;
  }
  float mean = block_sum(v[0] + v[1] + v[2], red) * (1.f / H_);
  float d0 = v[0] - mean, d1 = v[1] - mean, d2 = v[2] - mean;
  float var = block_sum(d0 * d0 + d1 * d1 + d2 * d2, red) * (1.f / H_);
  float rstd = rsqrtf(var + EPS_);
#pragma unroll
  for (int j = 0; j < 3; ++j) {
    int h = t + j * 256;
    relay[(long)b * H_ + h] = (v[j] - mean) * rstd * g[h] + bta[h];
  }
}

} // namespace

// ---------------------------------------------------------------------------
extern "C" void kernel_launch(void* const* d_in, const int* in_sizes, int n_in,
                              void* d_out, int out_size, void* d_ws, size_t ws_size,
                              hipStream_t stream) {
  (void)in_sizes; (void)n_in; (void)out_size; (void)ws_size;

  const float* data        = (const float*)d_in[0];
  const int*   mask        = (const int*)  d_in[1];
  const int*   edge_matrix = (const int*)  d_in[2];
  const int*   edge_label  = (const int*)  d_in[3];
  const float* ln_g        = (const float*)d_in[4];
  const float* ln_b        = (const float*)d_in[5];
  const float* r_wq        = (const float*)d_in[6];
  const float* r_bq        = (const float*)d_in[7];
  const float* r_wo        = (const float*)d_in[8];
  const float* r_bo        = (const float*)d_in[9];
  const float* r_emb       = (const float*)d_in[10];
  const float* r_we        = (const float*)d_in[11];
  const float* r_bwe       = (const float*)d_in[12];
  const float* r_wedge     = (const float*)d_in[13];
  const float* r_bwedge    = (const float*)d_in[14];
  const float* s_wq        = (const float*)d_in[15];
  const float* s_bq        = (const float*)d_in[16];
  const float* s_wk        = (const float*)d_in[17];
  const float* s_bk        = (const float*)d_in[18];
  const float* s_wo        = (const float*)d_in[19];
  const float* s_bo        = (const float*)d_in[20];

  float* nodes = (float*)d_out;                    // [B,L,H]
  float* relay = nodes + (long)B_ * L_ * H_;       // [B,H]

  // workspace carve-out
  char*  w   = (char*)d_ws;
  size_t off = 0;
  auto alloc = [&](size_t bytes) -> void* {
    void* p = w + off;
    off += (bytes + 255) & ~(size_t)255;
    return p;
  };
  const long WSZ = (long)IT_ * C_ * H_;
  __bf16* wq_bf   = (__bf16*)alloc(WSZ * 2);
  __bf16* wo_bf   = (__bf16*)alloc(WSZ * 2);
  __bf16* wk_bf   = (__bf16*)alloc(WSZ * 2);
  __bf16* xn_bf   = (__bf16*)alloc((long)B_ * L_ * C_ * 2);
  float*  q_f32   = (float*) alloc((long)B_ * L_ * C_ * 4);
  __bf16* q_bf    = (__bf16*)alloc((long)B_ * L_ * C_ * 2);
  float*  s1      = (float*) alloc((long)B_ * NH_ * L_ * 4);
  float*  s2      = (float*) alloc((long)B_ * NH_ * L_ * 4);
  float*  Ttab    = (float*) alloc((long)NLAB_ * NH_ * 4);
  __bf16* alpha   = (__bf16*)alloc((long)B_ * NH_ * L_ * L_ * 2);
  __bf16* att2t   = (__bf16*)alloc((long)B_ * L_ * C_ * 2);
  float*  ret     = (float*) alloc((long)B_ * L_ * H_ * 4);
  __bf16* y_bf    = (__bf16*)alloc((long)B_ * LP1_ * C_ * 2);
  float*  kk      = (float*) alloc((long)B_ * LP1_ * C_ * 4);
  float*  qs      = (float*) alloc((long)B_ * C_ * 4);
  float*  soutp   = (float*) alloc((long)B_ * C_ * 4);

  // ---- init: nodes = data ; relay = mean_l(data) ; weights -> bf16 ----
  {
    long n = (long)B_ * L_ * H_;
    copy_f32_kernel<<<(unsigned)((n + 255) / 256), 256, 0, stream>>>(data, nodes, n);
    relay_init_kernel<<<(B_ * H_ + 255) / 256, 256, 0, stream>>>(data, relay);
    conv_bf16_kernel<<<(unsigned)((WSZ + 255) / 256), 256, 0, stream>>>(r_wq, wq_bf, WSZ);
    conv_bf16_kernel<<<(unsigned)((WSZ + 255) / 256), 256, 0, stream>>>(r_wo, wo_bf, WSZ);
    conv_bf16_kernel<<<(unsigned)((WSZ + 255) / 256), 256, 0, stream>>>(s_wk, wk_bf, WSZ);
  }

  for (int i = 0; i < IT_; ++i) {
    const float* g   = ln_g + (long)i * H_;
    const float* bb  = ln_b + (long)i * H_;
    const __bf16* wq_i = wq_bf + (long)i * C_ * H_;
    const __bf16* wo_i = wo_bf + (long)i * H_ * C_;
    const __bf16* wk_i = wk_bf + (long)i * C_ * H_;
    float invtemp = 1.0f / sqrtf((float)(i + 1));

    // T table (fold r_we, r_bwe, r_emb with w3)
    table_kernel<<<1, 256, 0, stream>>>(r_we + (long)i * C_ * HD_, r_bwe + (long)i * C_,
                                        r_emb + (long)i * NLAB_ * HD_, r_wedge + (long)i * 3 * HD_,
                                        Ttab);
    // xn = LN(nodes) -> bf16
    ln_bf_kernel<<<B_ * L_, 256, 0, stream>>>(nodes, g, bb, xn_bf);
    // q = xn @ wq^T + bq  (f32 + bf16)
    gemm_awt_kernel<<<dim3(6, 32, 1), 256, 0, stream>>>(
        xn_bf, 0, wq_i, r_bq + (long)i * C_, q_f32, 0, q_bf, 0,
        B_ * L_, C_, H_, C_);
    // s1, s2
    s12_kernel<<<(B_ * NH_ * L_ + 255) / 256, 256, 0, stream>>>(
        q_f32, r_wedge + (long)i * 3 * HD_, s1, s2);
    // masked additive-attention softmax -> alpha (bf16, temperature applied)
    edge_softmax_kernel<<<B_ * L_, 256, 0, stream>>>(
        s1, s2, Ttab, edge_matrix, edge_label, r_bwedge + i, invtemp, alpha);
    // att = alpha @ q_head -> scrambled att2t (bf16)
    gemm_att_kernel<<<dim3(1, 4, B_ * NH_), 256, 0, stream>>>(alpha, q_bf, att2t);
    // ret = att2t @ wo^T + bo (batched over b)
    gemm_awt_kernel<<<dim3(6, 4, B_), 256, 0, stream>>>(
        att2t, (long)L_ * C_, wo_i, r_bo + (long)i * H_, ret, (long)L_ * H_, nullptr, 0,
        L_, C_, C_, H_);
    // nodes = LN(nodes + masked leaky(ret)); also fills y_bf rows 1..256
    ln_res_kernel<<<B_ * L_, 256, 0, stream>>>(nodes, ret, mask, g, bb, y_bf);
    // y_bf row 0 = relay
    relay_to_y_kernel<<<(B_ * C_ + 255) / 256, 256, 0, stream>>>(relay, y_bf);
    // kk = y @ s_wk^T + bk (batched over b, M=257)
    gemm_awt_kernel<<<dim3(6, 5, B_), 256, 0, stream>>>(
        y_bf, (long)LP1_ * C_, wk_i, s_bk + (long)i * C_, kk, (long)LP1_ * C_, nullptr, 0,
        LP1_, C_, H_, C_);
    // qs = relay @ s_wq^T + bq
    qs_kernel<<<(B_ * C_ + 255) / 256, 256, 0, stream>>>(
        relay, s_wq + (long)i * C_ * H_, s_bq + (long)i * C_, qs);
    // star attention heads
    star_kernel<<<B_ * NH_, 256, 0, stream>>>(qs, kk, mask, soutp);
    // relay = LN(relay + leaky(soutp @ s_wo^T + bo))
    relay_update_kernel<<<B_, 256, 0, stream>>>(
        relay, soutp, s_wo + (long)i * H_ * C_, s_bo + (long)i * H_, g, bb);
  }
}
